// MultiHeadAttention_3367254360519
// MI455X (gfx1250) — matrix operand order repaired and checked by reading, not compiled, not run on hardware
//
#include <hip/hip_runtime.h>
#include <hip/hip_bf16.h>

#define MODEL_DIM 1024
#define NUM_HEADS 16
#define HEAD_DIM  64
#define SEQ       2048
#define BATCH     2
#define MTOT      (BATCH * SEQ)   // 4096

typedef __attribute__((ext_vector_type(16))) __bf16   v16bf;
typedef __attribute__((ext_vector_type(8)))  __bf16   v8bf;
typedef __attribute__((ext_vector_type(4)))  __bf16   v4bf;
typedef __attribute__((ext_vector_type(8)))  float    v8f;
typedef __attribute__((ext_vector_type(4)))  float    v4f;
typedef __attribute__((ext_vector_type(4)))  unsigned v4u;
typedef __attribute__((ext_vector_type(8)))  unsigned v8u;

// ---------------- CDNA5 primitives ----------------

// D = A(16x32) * B(32x16) + C, f32 accumulate
static __device__ __forceinline__ v8f wmma_bf16(v16bf a, v16bf b, v8f c) {
  return __builtin_amdgcn_wmma_f32_16x16x32_bf16(false, a, false, b, (short)0, c,
                                                 false, false);
}

// Async DMA: 16 bytes per lane, global -> LDS, tracked by ASYNCcnt.
static __device__ __forceinline__ void async_load_b128(void* lds, const void* gptr) {
  asm volatile("global_load_async_to_lds_b128 %0, %1, off"
               :: "v"((unsigned)(uintptr_t)lds), "v"(gptr)
               : "memory");
}
static __device__ __forceinline__ void wait_asynccnt0() {
  asm volatile("s_wait_asynccnt 0x0" ::: "memory");
}

// Tensor Data Mover: DMA a 2D tile (tile_w x tile_h elements, 2-byte elements,
// row stride = stride_elems) from global into LDS (rows compacted).
// Descriptor per CDNA5 ISA (groups 2/3 = NULL -> 2D form).
static __device__ __forceinline__ void tdm_load_2d_bf16(void* lds, const void* gptr,
                                                        unsigned tile_w, unsigned tile_h,
                                                        unsigned tensor_w, unsigned tensor_h,
                                                        unsigned stride_elems) {
  const unsigned long long ga = (unsigned long long)(uintptr_t)gptr;
  v4u g0;
  g0[0] = 1u;                                    // count=1, user descriptor
  g0[1] = (unsigned)(uintptr_t)lds;              // lds_addr
  g0[2] = (unsigned)(ga & 0xffffffffu);          // global_addr[31:0]
  g0[3] = (unsigned)((ga >> 32) & 0x01ffffffu) | 0x80000000u; // addr[56:32] | type=2
  v8u g1;
  g1[0] = 0x00010000u;                           // data_size=1 (2B); no pad/iter/mask
  g1[1] = (tensor_w & 0xffffu) << 16;            // tensor_dim0[15:0] (bits 63:48)
  g1[2] = (tensor_w >> 16) | ((tensor_h & 0xffffu) << 16); // dim0 hi | dim1 lo
  g1[3] = (tensor_h >> 16) | (tile_w << 16);     // dim1 hi | tile_dim0
  g1[4] = tile_h;                                // tile_dim1, tile_dim2 = 0
  g1[5] = stride_elems;                          // tensor_dim0_stride[31:0]
  g1[6] = 0u;                                    // stride0 hi | stride1 lo
  g1[7] = 0u;
  asm volatile("tensor_load_to_lds %0, %1" :: "s"(g0), "s"(g1) : "memory");
}

// ---------------- fragment loaders (ISA §7.12.2 layouts) ----------------

// A fragment (16x32): lane holds row0+(lane&15); elems 0..7 -> K=k0+8*(lane>>4)+i,
// elems 8..15 -> K=k0+16+8*(lane>>4)+i
static __device__ __forceinline__ v16bf load_afrag(const __bf16* base, int row0,
                                                   int k0, int ld, int lane) {
  const __bf16* p = base + (size_t)(row0 + (lane & 15)) * ld + k0 + ((lane >> 4) << 3);
  v8bf lo = *(const v8bf*)p;
  v8bf hi = *(const v8bf*)(p + 16);
  v16bf r;
#pragma unroll
  for (int i = 0; i < 8; ++i) { r[i] = lo[i]; r[8 + i] = hi[i]; }
  return r;
}

// B fragment (32x16) from [n][k] row-major source: lane holds column n0+(lane&15);
// elems 0..7 -> K=k0+8*(lane>>4)+i, elems 8..15 -> K=k0+16+8*(lane>>4)+i
static __device__ __forceinline__ v16bf load_bfrag(const __bf16* base, int n0,
                                                   int k0, int ld, int lane) {
  const __bf16* p = base + (size_t)(n0 + (lane & 15)) * ld + k0 + ((lane >> 4) << 3);
  v8bf lo = *(const v8bf*)p;
  v8bf hi = *(const v8bf*)(p + 16);
  v16bf r;
#pragma unroll
  for (int i = 0; i < 8; ++i) { r[i] = lo[i]; r[8 + i] = hi[i]; }
  return r;
}

__global__ void cast_f32_bf16(const float* __restrict__ in, __bf16* __restrict__ out,
                              int n4) {
  int i = blockIdx.x * blockDim.x + threadIdx.x;
  if (i < n4) {
    v4f v = ((const v4f*)in)[i];
    v4bf o;
    o[0] = (__bf16)v[0]; o[1] = (__bf16)v[1];
    o[2] = (__bf16)v[2]; o[3] = (__bf16)v[3];
    ((v4bf*)out)[i] = o;
  }
}

// ---------------- GEMM: C = A[MxK] * W[NxK]^T ----------------
// TDM double-buffered K-panels of 64. Epilogue modes:
// 0: bf16 out[m][n]   1: bf16 [bh][s][hd]   2: bf16 V^T [bh][hd][s]
// 3: f32 out[m][n] + bias[n]
__global__ void __launch_bounds__(256)
gemm_bf16(const __bf16* __restrict__ A, const __bf16* __restrict__ W,
          void* __restrict__ outp, const float* __restrict__ bias,
          int N, int Kd, int mode) {
  __shared__ __attribute__((aligned(16))) __bf16 As[2][128 * 64];
  __shared__ __attribute__((aligned(16))) __bf16 Ws[2][64 * 64];

  const int tid  = threadIdx.x;
  const int lane = tid & 31;
  const int wave = tid >> 5;
  const int wm = wave & 3, wn = wave >> 2;   // 4x2 wave grid, 32x32 per wave
  const int m0 = blockIdx.y * 128;
  const int n0 = blockIdx.x * 64;

  v8f acc[2][2];
#pragma unroll
  for (int i = 0; i < 2; ++i)
#pragma unroll
    for (int j = 0; j < 2; ++j) { v8f z = {}; acc[i][j] = z; }

  // TDM issue: wave 0 stages the A panel, wave 1 the W panel (EXEC-independent,
  // so it must be wave-guarded).
  auto stage = [&](int buf, int k0) {
    if (wave == 0)
      tdm_load_2d_bf16(&As[buf][0], A + (size_t)m0 * Kd + k0,
                       64, 128, (unsigned)Kd, MTOT, (unsigned)Kd);
    else if (wave == 1)
      tdm_load_2d_bf16(&Ws[buf][0], W + (size_t)n0 * Kd + k0,
                       64, 64, (unsigned)Kd, (unsigned)N, (unsigned)Kd);
  };

  const int nk = Kd / 64;
  stage(0, 0);
  for (int kt = 0; kt < nk; ++kt) {
    const int cur = kt & 1;
    __builtin_amdgcn_s_wait_tensorcnt(0);  // issuing waves wait for their DMA
    __syncthreads();                       // tile visible to (and buffers free for) all
    if (kt + 1 < nk) stage(cur ^ 1, (kt + 1) * 64);

#pragma unroll
    for (int kk = 0; kk < 2; ++kk) {
      v16bf af[2], bfm[2];
#pragma unroll
      for (int i = 0; i < 2; ++i)
        af[i] = load_afrag(&As[cur][0], wm * 32 + i * 16, kk * 32, 64, lane);
#pragma unroll
      for (int j = 0; j < 2; ++j)
        bfm[j] = load_bfrag(&Ws[cur][0], wn * 32 + j * 16, kk * 32, 64, lane);
#pragma unroll
      for (int i = 0; i < 2; ++i)
#pragma unroll
        for (int j = 0; j < 2; ++j) acc[i][j] = wmma_bf16(af[i], bfm[j], acc[i][j]);
    }
  }

#pragma unroll
  for (int i = 0; i < 2; ++i) {
#pragma unroll
    for (int j = 0; j < 2; ++j) {
      const int nn = n0 + wn * 32 + j * 16 + (lane & 15);
#pragma unroll
      for (int r = 0; r < 8; ++r) {
        const int mm = m0 + wm * 32 + i * 16 + r + ((lane >> 4) << 3);
        const float val = acc[i][j][r];
        if (mode == 0) {
          ((__bf16*)outp)[(size_t)mm * N + nn] = (__bf16)val;
        } else if (mode == 1) {
          const int b = mm >> 11, s = mm & (SEQ - 1);
          const int h = nn >> 6,  hd = nn & 63;
          ((__bf16*)outp)[(((size_t)(b * NUM_HEADS + h) * SEQ + s) << 6) + hd] = (__bf16)val;
        } else if (mode == 2) {
          const int b = mm >> 11, s = mm & (SEQ - 1);
          const int h = nn >> 6,  hd = nn & 63;
          ((__bf16*)outp)[((size_t)(b * NUM_HEADS + h) * HEAD_DIM + hd) * SEQ + s] = (__bf16)val;
        } else {
          ((float*)outp)[(size_t)mm * N + nn] = val + bias[nn];
        }
      }
    }
  }
}

// ---------------- Flash attention ----------------
// grid (S/64, B*H), 128 threads (4 waves, 16 q-rows each).
// Q/K: [bh][s][hd]  Vt: [bh][hd][s]  -> Abuf bf16 [b*S+s][h*64+hd]
// K/V tiles double-buffered via global_load_async_to_lds_b128 (ASYNCcnt).
__global__ void __launch_bounds__(128)
flash_attn(const __bf16* __restrict__ Q, const __bf16* __restrict__ Kg,
           const __bf16* __restrict__ Vt, __bf16* __restrict__ Abuf) {
  __shared__ __attribute__((aligned(16))) __bf16 Ks[2][64 * 64];
  __shared__ __attribute__((aligned(16))) __bf16 Vs[2][64 * 64];    // [hd][kv]
  __shared__ __attribute__((aligned(16))) __bf16 Ps[4][16 * 64];    // per-wave P

  const int tid  = threadIdx.x;
  const int lane = tid & 31;
  const int wave = tid >> 5;
  const int bh = blockIdx.y;
  const int b = bh >> 4, h = bh & 15;
  const int sq0 = blockIdx.x * 64;

  const __bf16* Qp = Q  + ((size_t)bh * SEQ + sq0 + wave * 16) * HEAD_DIM;
  const __bf16* Kp = Kg + (size_t)bh * SEQ * HEAD_DIM;
  const __bf16* Vp = Vt + (size_t)bh * HEAD_DIM * SEQ;

  const v16bf qf0 = load_afrag(Qp, 0, 0, HEAD_DIM, lane);
  const v16bf qf1 = load_afrag(Qp, 0, 32, HEAD_DIM, lane);

  float mrow[8], lrow[8];
  v8f of[4];
#pragma unroll
  for (int r = 0; r < 8; ++r) { mrow[r] = -1e30f; lrow[r] = 0.f; }
#pragma unroll
  for (int f = 0; f < 4; ++f) { v8f z = {}; of[f] = z; }

  const int ldr = tid >> 1;          // 0..63
  const int ldc = (tid & 1) << 5;    // 0 / 32

  auto stage = [&](int buf, int kv0) {
    const __bf16* ks = Kp + (size_t)(kv0 + ldr) * HEAD_DIM + ldc;
    const __bf16* vsrc = Vp + (size_t)ldr * SEQ + kv0 + ldc;
#pragma unroll
    for (int c = 0; c < 4; ++c) {
      async_load_b128(&Ks[buf][ldr * 64 + ldc + c * 8], ks + c * 8);
      async_load_b128(&Vs[buf][ldr * 64 + ldc + c * 8], vsrc + c * 8);
    }
  };

  const int nt = SEQ / 64;
  stage(0, 0);
  for (int kt = 0; kt < nt; ++kt) {
    const int cur = kt & 1;
    wait_asynccnt0();                 // own async stores to LDS complete
    __syncthreads();                  // whole tile visible; prev buffer free
    if (kt + 1 < nt) stage(cur ^ 1, (kt + 1) * 64);

    // S = (Q K^T) * 1/sqrt(64) : 16 x 64 per wave
    v8f sf[4];
#pragma unroll
    for (int nf = 0; nf < 4; ++nf) {
      v8f c = {};
      c = wmma_bf16(qf0, load_bfrag(&Ks[cur][0], nf * 16, 0, 64, lane), c);
      c = wmma_bf16(qf1, load_bfrag(&Ks[cur][0], nf * 16, 32, 64, lane), c);
      sf[nf] = c * 0.125f;
    }

    // online softmax (lane = column, VGPR = row, lane-half = row+8)
    __bf16* Pw = &Ps[wave][0];
#pragma unroll
    for (int r = 0; r < 8; ++r) {
      float v = sf[0][r];
#pragma unroll
      for (int nf = 1; nf < 4; ++nf) v = fmaxf(v, sf[nf][r]);
      v = fmaxf(v, __shfl_xor(v, 1, 32));
      v = fmaxf(v, __shfl_xor(v, 2, 32));
      v = fmaxf(v, __shfl_xor(v, 4, 32));
      v = fmaxf(v, __shfl_xor(v, 8, 32));
      const float mn = fmaxf(mrow[r], v);
      const float alpha = __expf(mrow[r] - mn);
      mrow[r] = mn;
      float sum = 0.f;
#pragma unroll
      for (int nf = 0; nf < 4; ++nf) {
        const float p = __expf(sf[nf][r] - mn);
        sf[nf][r] = p;
        sum += p;
      }
      sum += __shfl_xor(sum, 1, 32);
      sum += __shfl_xor(sum, 2, 32);
      sum += __shfl_xor(sum, 4, 32);
      sum += __shfl_xor(sum, 8, 32);
      lrow[r] = lrow[r] * alpha + sum;
#pragma unroll
      for (int f = 0; f < 4; ++f) of[f][r] *= alpha;
      const int prow = r + ((lane >> 4) << 3);
#pragma unroll
      for (int nf = 0; nf < 4; ++nf)
        Pw[prow * 64 + nf * 16 + (lane & 15)] = (__bf16)sf[nf][r];
    }

    // O += P * V (P round-trips through per-wave LDS: D-layout -> A-layout)
    const v16bf pa0 = load_afrag(Pw, 0, 0, 64, lane);
    const v16bf pa1 = load_afrag(Pw, 0, 32, 64, lane);
#pragma unroll
    for (int f = 0; f < 4; ++f) {
      of[f] = wmma_bf16(pa0, load_bfrag(&Vs[cur][0], f * 16, 0, 64, lane), of[f]);
      of[f] = wmma_bf16(pa1, load_bfrag(&Vs[cur][0], f * 16, 32, 64, lane), of[f]);
    }
  }

#pragma unroll
  for (int f = 0; f < 4; ++f) {
    const int d = h * HEAD_DIM + f * 16 + (lane & 15);
#pragma unroll
    for (int r = 0; r < 8; ++r) {
      const int row = r + ((lane >> 4) << 3);
      const size_t m = (size_t)b * SEQ + sq0 + wave * 16 + row;
      Abuf[m * MODEL_DIM + d] = (__bf16)(of[f][r] / lrow[r]);
    }
  }
}

extern "C" void kernel_launch(void* const* d_in, const int* in_sizes, int n_in,
                              void* d_out, int out_size, void* d_ws, size_t ws_size,
                              hipStream_t stream) {
  const float* x  = (const float*)d_in[0];
  const float* Wq = (const float*)d_in[1];
  const float* Wk = (const float*)d_in[2];
  const float* Wv = (const float*)d_in[3];
  const float* Wo = (const float*)d_in[4];
  const float* bo = (const float*)d_in[5];

  char* ws = (char*)d_ws;
  __bf16* xb  = (__bf16*)(ws);                            // 8 MB
  __bf16* Wqb = (__bf16*)(ws + (size_t)(8)  * (1 << 20));
  __bf16* Wkb = (__bf16*)(ws + (size_t)(10) * (1 << 20));
  __bf16* Wvb = (__bf16*)(ws + (size_t)(12) * (1 << 20));
  __bf16* Wob = (__bf16*)(ws + (size_t)(14) * (1 << 20));
  __bf16* Qb  = (__bf16*)(ws + (size_t)(16) * (1 << 20)); // 8 MB
  __bf16* Kb  = (__bf16*)(ws + (size_t)(24) * (1 << 20)); // 8 MB
  __bf16* Vb  = (__bf16*)(ws + (size_t)(32) * (1 << 20)); // 8 MB (transposed)
  __bf16* Ab  = (__bf16*)(ws + (size_t)(40) * (1 << 20)); // 8 MB

  const int nx4 = (MTOT * MODEL_DIM) / 4;
  const int nw4 = (MODEL_DIM * MODEL_DIM) / 4;
  cast_f32_bf16<<<(nx4 + 255) / 256, 256, 0, stream>>>(x,  xb,  nx4);
  cast_f32_bf16<<<(nw4 + 255) / 256, 256, 0, stream>>>(Wq, Wqb, nw4);
  cast_f32_bf16<<<(nw4 + 255) / 256, 256, 0, stream>>>(Wk, Wkb, nw4);
  cast_f32_bf16<<<(nw4 + 255) / 256, 256, 0, stream>>>(Wv, Wvb, nw4);
  cast_f32_bf16<<<(nw4 + 255) / 256, 256, 0, stream>>>(Wo, Wob, nw4);

  dim3 gg(MODEL_DIM / 64, MTOT / 128);  // (16, 32)
  gemm_bf16<<<gg, 256, 0, stream>>>(xb, Wqb, Qb, nullptr, MODEL_DIM, MODEL_DIM, 1);
  gemm_bf16<<<gg, 256, 0, stream>>>(xb, Wkb, Kb, nullptr, MODEL_DIM, MODEL_DIM, 1);
  gemm_bf16<<<gg, 256, 0, stream>>>(xb, Wvb, Vb, nullptr, MODEL_DIM, MODEL_DIM, 2);

  flash_attn<<<dim3(SEQ / 64, BATCH * NUM_HEADS), 128, 0, stream>>>(Qb, Kb, Vb, Ab);

  gemm_bf16<<<gg, 256, 0, stream>>>(Ab, Wob, d_out, bo, MODEL_DIM, MODEL_DIM, 3);
}
